// TimeSeriesTransformer_65438121722682
// MI455X (gfx1250) — matrix operand order repaired and checked by reading, not compiled, not run on hardware
//
#include <hip/hip_runtime.h>
#include <hip/hip_bf16.h>

// ---------------------------------------------------------------------------
// Informer ProbSparse attention for MI455X (gfx1250, wave32, WMMA).
// B=4, L=2048, E=512, H=8, Dh=64, u = n_top = 40 (padded to 48 for 16x tiles).
// GEMMs: bf16 inputs pre-converted once, v_wmma_f32_16x16x32_bf16 accumulate
// in f32.  Softmax / top-k / probe metric / mean stay in f32.
// ---------------------------------------------------------------------------

#define BB 4
#define LL 2048
#define EE 512
#define HH 8
#define DH 64
#define UU 40
#define UP 48   // padded top-k rows
#define M8 (BB * LL)   // 8192 GEMM rows

typedef __attribute__((ext_vector_type(16))) __bf16 v16bf;
typedef __attribute__((ext_vector_type(8)))  float  v8f;
typedef __attribute__((ext_vector_type(4)))  float  f4;
typedef __attribute__((ext_vector_type(2)))  float  f2;
typedef __attribute__((ext_vector_type(8)))  unsigned short us8;
typedef __attribute__((ext_vector_type(4)))  unsigned short us4;

union BF16x16 { v16bf v; unsigned short u[16]; us8 h[2]; };
union F4x1    { f4 v; float f[4]; };

__device__ __forceinline__ unsigned short f2bf(float f) {
  unsigned int x = __float_as_uint(f);
  x += 0x7fffu + ((x >> 16) & 1u);          // round-to-nearest-even
  return (unsigned short)(x >> 16);
}

__device__ __forceinline__ v8f wmma_bf16(v16bf a, v16bf b, v8f c) {
  return __builtin_amdgcn_wmma_f32_16x16x32_bf16(
      /*neg_a=*/false, a, /*neg_b=*/false, b,
      /*c_mod=*/(short)0, c, /*reuse_a=*/false, /*reuse_b=*/false);
}

// A-matrix (16x32 bf16, MxK) from bf16 row-major [.,ld].
// Lane: M=lane&15; 16 contiguous K at k0 + 16*(lane>=16) (matches loadB_bf).
__device__ __forceinline__ v16bf loadA_bf(const unsigned short* X, int ld, int m0, int k0, int lane) {
  int m = m0 + (lane & 15), half = lane >> 4;
  const unsigned short* p0 = X + (size_t)m * ld + k0 + 16 * half;
  BF16x16 r;
  r.h[0] = *(const us8*)(p0);
  r.h[1] = *(const us8*)(p0 + 8);
  return r.v;
}

// B-matrix (32x16 bf16, KxN) where B[k][n] = W[n][k] (W row-major [N,ld]).
// Lane: n = n0+(lane&15); 16 contiguous K at k0 + 16*(lane>=16).
__device__ __forceinline__ v16bf loadB_bf(const unsigned short* W, int ld, int n0, int k0, int lane) {
  const unsigned short* p = W + (size_t)(n0 + (lane & 15)) * ld + k0 + 16 * (lane >> 4);
  BF16x16 r;
  r.h[0] = *(const us8*)(p);
  r.h[1] = *(const us8*)(p + 8);
  return r.v;
}

// ---------------------------------------------------------------------------
// Elementwise f32 -> bf16 (4 per thread)
// ---------------------------------------------------------------------------
__global__ void __launch_bounds__(256)
cvt_bf16(const float* __restrict__ x, unsigned short* __restrict__ y) {
  size_t i = (size_t)blockIdx.x * 256 + threadIdx.x;
  F4x1 v; v.v = ((const f4*)x)[i];
  us4 o;
  o.x = f2bf(v.f[0]); o.y = f2bf(v.f[1]); o.z = f2bf(v.f[2]); o.w = f2bf(v.f[3]);
  ((us4*)y)[i] = o;
}

// ---------------------------------------------------------------------------
// GEMM: Y[8192,512] = X[8192,512] @ W[512,512]^T + bias, bf16 in / f32 out,
// optional bf16 copy.  One wave per 64x32 tile (4 M-subtiles x 2 N-subtiles),
// 8 waves/block.  8 WMMAs + 12x16B loads per 32-K step.
// ---------------------------------------------------------------------------
__global__ void __launch_bounds__(256)
gemm_bf16(const unsigned short* __restrict__ X, const unsigned short* __restrict__ W,
          const float* __restrict__ bias, float* __restrict__ Y,
          unsigned short* __restrict__ Ybf) {
  int wave = blockIdx.x * 8 + (threadIdx.x >> 5);
  int lane = threadIdx.x & 31;
  const int tilesM = M8 / 64;                 // 128
  int tm = (wave % tilesM) * 64;
  int tn = (wave / tilesM) * 32;
  v8f acc[4][2] = {};
#pragma unroll 2
  for (int k0 = 0; k0 < EE; k0 += 32) {
    if (k0 + 32 < EE) {
      __builtin_prefetch(X + (size_t)(tm + (lane & 15)) * EE + k0 + 32, 0, 1);
      __builtin_prefetch(W + (size_t)(tn + (lane & 15)) * EE + k0 + 32, 0, 1);
    }
    v16bf b0 = loadB_bf(W, EE, tn,      k0, lane);
    v16bf b1 = loadB_bf(W, EE, tn + 16, k0, lane);
#pragma unroll
    for (int im = 0; im < 4; ++im) {
      v16bf a = loadA_bf(X, EE, tm + im * 16, k0, lane);
      acc[im][0] = wmma_bf16(a, b0, acc[im][0]);
      acc[im][1] = wmma_bf16(a, b1, acc[im][1]);
    }
  }
  int half = lane >> 4;
  float bn0 = bias[tn + (lane & 15)];
  float bn1 = bias[tn + 16 + (lane & 15)];
#pragma unroll
  for (int im = 0; im < 4; ++im)
#pragma unroll
    for (int in = 0; in < 2; ++in)
#pragma unroll
      for (int rr = 0; rr < 8; ++rr) {
        int m = tm + im * 16 + rr + 8 * half;
        int n = tn + in * 16 + (lane & 15);
        float v = acc[im][in][rr] + (in ? bn1 : bn0);
        Y[(size_t)m * EE + n] = v;
        if (Ybf) Ybf[(size_t)m * EE + n] = f2bf(v);
      }
}

// ---------------------------------------------------------------------------
// Sampled sparsity metric  M[b,h,l] = max_s(q.k_s) - sum_s(q.k_s)/L
// One wave per (b,h,l); lane holds 2 of 64 dims; wave32 xor-reduce.
// ---------------------------------------------------------------------------
__global__ void __launch_bounds__(256)
probe_kernel(const float* __restrict__ q, const float* __restrict__ kf,
             const int* __restrict__ idxs, float* __restrict__ Mout) {
  int g = blockIdx.x * 8 + (threadIdx.x >> 5);
  int lane = threadIdx.x & 31;
  int b = g >> 14, h = (g >> 11) & 7, l = g & 2047;
  const float* qb = q + ((size_t)(b * LL + l)) * EE + h * DH;
  f2 qv = *(const f2*)(qb + 2 * lane);
  float mx = -3.4e38f, sm = 0.f;
  for (int s = 0; s < UU; ++s) {
    int ki = idxs[l * UU + s];
    const float* kb = kf + ((size_t)(b * LL + ki)) * EE + h * DH;
    f2 kv = *(const f2*)(kb + 2 * lane);
    float d = qv.x * kv.x + qv.y * kv.y;
#pragma unroll
    for (int msk = 16; msk > 0; msk >>= 1) d += __shfl_xor(d, msk, 32);
    mx = fmaxf(mx, d);
    sm += d;
  }
  if (lane == 0) Mout[g] = mx - sm * (1.0f / (float)LL);
}

// ---------------------------------------------------------------------------
// Top-40 indices of M[b,h,:] via iterative block argmax in LDS.
// ---------------------------------------------------------------------------
__global__ void __launch_bounds__(256)
topk_kernel(const float* __restrict__ Min, int* __restrict__ mtop) {
  __shared__ float sm[LL];
  __shared__ float bv[256];
  __shared__ int   bi[256];
  int bh = blockIdx.x, t = threadIdx.x;
  const float* row = Min + (size_t)bh * LL;
  for (int i = t; i < LL; i += 256) sm[i] = row[i];
  __syncthreads();
  for (int it = 0; it < UU; ++it) {
    float best = -3.4e38f; int besti = 0;
    for (int i = t; i < LL; i += 256) {
      float v = sm[i];
      if (v > best) { best = v; besti = i; }
    }
    bv[t] = best; bi[t] = besti;
    __syncthreads();
    for (int s = 128; s > 0; s >>= 1) {
      if (t < s && bv[t + s] > bv[t]) { bv[t] = bv[t + s]; bi[t] = bi[t + s]; }
      __syncthreads();
    }
    if (t == 0) { mtop[bh * UU + it] = bi[0]; sm[bi[0]] = -3.4e38f; }
    __syncthreads();
  }
}

// vmean[b,h,d] = mean_l v[b,l,h,d]
__global__ void __launch_bounds__(256)
vmean_kernel(const float* __restrict__ vf, float* __restrict__ vmean) {
  int bh = blockIdx.x, b = bh >> 3, h = bh & 7;
  int t = threadIdx.x, d = t & 63, seg = t >> 6;
  float s = 0.f;
  for (int l = seg * 512; l < seg * 512 + 512; ++l)
    s += vf[((size_t)(b * LL + l)) * EE + h * DH + d];
  __shared__ float red[256];
  red[t] = s;
  __syncthreads();
  if (seg == 0)
    vmean[bh * DH + d] =
        (red[d] + red[64 + d] + red[128 + d] + red[192 + d]) * (1.0f / (float)LL);
}

// ctx[b,l,h,d] = vmean[b,h,d]  (broadcast fill before scatter)
__global__ void __launch_bounds__(256)
fill_ctx(float* __restrict__ ctx, const float* __restrict__ vmean) {
  size_t i = (size_t)blockIdx.x * 256 + threadIdx.x;   // < B*L*E
  int e = (int)(i & (EE - 1));
  int b = (int)(i >> 9) >> 11;
  ctx[i] = vmean[((b << 3) + (e >> 6)) * DH + (e & 63)];
}

// Gather top-40 query rows (zero-padded to 48) into bf16 tile.
__global__ void __launch_bounds__(256)
gather_qtop(const float* __restrict__ q, const int* __restrict__ mtop,
            unsigned short* __restrict__ qtop) {
  int i = blockIdx.x * 256 + threadIdx.x;              // < 32*48*64
  int d = i & 63, u = (i >> 6) % UP, bh = i / (UP * DH);
  int b = bh >> 3, h = bh & 7;
  float val = 0.f;
  if (u < UU) {
    int l = mtop[bh * UU + u];
    val = q[((size_t)(b * LL + l)) * EE + h * DH + d];
  }
  qtop[i] = f2bf(val);
}

// ---------------------------------------------------------------------------
// S[bh][48][2048] = Qtop(48x64) @ K^T  — WMMA; B-tiles are contiguous bf16
// key rows.
// ---------------------------------------------------------------------------
__global__ void __launch_bounds__(256)
scores_kernel(const unsigned short* __restrict__ qtop,
              const unsigned short* __restrict__ kbf, float* __restrict__ S) {
  int wave = blockIdx.x * 8 + (threadIdx.x >> 5);      // < 32*384
  int lane = threadIdx.x & 31;
  int bh = wave / 384, r = wave % 384;
  int m0 = (r % 3) * 16, n0 = (r / 3) * 16;
  int b = bh >> 3, h = bh & 7;
  const unsigned short* A = qtop + (size_t)bh * UP * DH;
  const unsigned short* krow =
      kbf + ((size_t)(b * LL + n0)) * EE + h * DH;
  v8f acc = {};
#pragma unroll
  for (int k0 = 0; k0 < DH; k0 += 32) {
    v16bf a = loadA_bf(A, DH, m0, k0, lane);
    v16bf bm = loadB_bf(krow, EE, 0, k0, lane);
    acc = wmma_bf16(a, bm, acc);
  }
  float* Srow = S + (size_t)bh * UP * LL;
  int half = lane >> 4, n = n0 + (lane & 15);
#pragma unroll
  for (int rr = 0; rr < 8; ++rr)
    Srow[(size_t)(m0 + rr + 8 * half) * LL + n] = acc[rr];
}

// f32 softmax per top row; emit bf16 probabilities.
__global__ void __launch_bounds__(256)
softmax_kernel(float* __restrict__ S, unsigned short* __restrict__ pbf) {
  int row = blockIdx.x;                                // < 32*40
  int bh = row / UU, u = row % UU;
  float* s = S + ((size_t)bh * UP + u) * LL;
  unsigned short* p = pbf + ((size_t)bh * UP + u) * LL;
  int t = threadIdx.x;
  __shared__ float red[256];
  float m = -3.4e38f;
  for (int i = t; i < LL; i += 256) m = fmaxf(m, s[i]);
  red[t] = m; __syncthreads();
  for (int st = 128; st > 0; st >>= 1) {
    if (t < st) red[t] = fmaxf(red[t], red[t + st]);
    __syncthreads();
  }
  m = red[0]; __syncthreads();
  float sum = 0.f;
  for (int i = t; i < LL; i += 256) { float e = __expf(s[i] - m); s[i] = e; sum += e; }
  red[t] = sum; __syncthreads();
  for (int st = 128; st > 0; st >>= 1) {
    if (t < st) red[t] += red[t + st];
    __syncthreads();
  }
  float inv = 1.0f / red[0];
  for (int i = t; i < LL; i += 256) p[i] = f2bf(s[i] * inv);
}

// ---------------------------------------------------------------------------
// ctx_top = P(48x2048) @ V(2048x64), scatter valid rows into ctx.
// ---------------------------------------------------------------------------
__global__ void __launch_bounds__(256)
pv_kernel(const unsigned short* __restrict__ pbf,
          const unsigned short* __restrict__ vbf,
          const int* __restrict__ mtop, float* __restrict__ ctx) {
  int wave = blockIdx.x * 8 + (threadIdx.x >> 5);      // < 32*12
  int lane = threadIdx.x & 31;
  int bh = wave / 12, r = wave % 12;
  int m0 = (r % 3) * 16, d0 = (r / 3) * 16;
  int b = bh >> 3, h = bh & 7;
  const unsigned short* P = pbf + (size_t)bh * UP * LL;
  v8f acc = {};
  for (int k0 = 0; k0 < LL; k0 += 32) {
    v16bf a = loadA_bf(P, LL, m0, k0, lane);
    // B[k=key][col=d]: stride-E gather over 16 consecutive keys.
    const unsigned short* vb =
        vbf + ((size_t)(b * LL + k0 + 16 * (lane >> 4))) * EE + h * DH + d0 + (lane & 15);
    BF16x16 bb;
#pragma unroll
    for (int i = 0; i < 16; ++i) bb.u[i] = vb[(size_t)i * EE];
    acc = wmma_bf16(a, bb.v, acc);
  }
  int half = lane >> 4, d = d0 + (lane & 15);
#pragma unroll
  for (int rr = 0; rr < 8; ++rr) {
    int u = m0 + rr + 8 * half;
    if (u < UU) {
      int l = mtop[bh * UU + u];
      ctx[((size_t)(b * LL + l)) * EE + h * DH + d] = acc[rr];
    }
  }
}

// ---------------------------------------------------------------------------
extern "C" void kernel_launch(void* const* d_in, const int* in_sizes, int n_in,
                              void* d_out, int out_size, void* d_ws, size_t ws_size,
                              hipStream_t stream) {
  (void)in_sizes; (void)n_in; (void)out_size; (void)ws_size;
  const float* query = (const float*)d_in[0];
  const float* key   = (const float*)d_in[1];
  const float* value = (const float*)d_in[2];
  const float* Wq = (const float*)d_in[3];  const float* bq = (const float*)d_in[4];
  const float* Wk = (const float*)d_in[5];  const float* bk = (const float*)d_in[6];
  const float* Wv = (const float*)d_in[7];  const float* bv = (const float*)d_in[8];
  const float* Wo = (const float*)d_in[9];  const float* bo = (const float*)d_in[10];
  const int* idxs = (const int*)d_in[11];
  float* out = (float*)d_out;

  // Workspace layout (~140 MB, all 16B aligned)
  const size_t NQKV = (size_t)BB * LL * EE;            // 4,194,304
  const size_t NW   = (size_t)EE * EE;                 // 262,144
  float* ws    = (float*)d_ws;
  float* q     = ws;
  float* kf    = q + NQKV;
  float* vf    = kf + NQKV;
  float* ctx   = vf + NQKV;
  float* Mbuf  = ctx + NQKV;                           // B*H*L = 65536
  float* vmean = Mbuf + (size_t)BB * HH * LL;          // 2048
  float* S     = vmean + (size_t)BB * HH * DH;         // 32*48*2048
  unsigned short* kbf   = (unsigned short*)(S + (size_t)BB * HH * UP * LL);
  unsigned short* vbf   = kbf + NQKV;
  unsigned short* qtop  = vbf + NQKV;                  // 32*48*64
  unsigned short* pbf   = qtop + (size_t)BB * HH * UP * DH;
  unsigned short* xq    = pbf + (size_t)BB * HH * UP * LL;
  unsigned short* xk    = xq + NQKV;
  unsigned short* xv    = xk + NQKV;
  unsigned short* ctxbf = xv + NQKV;
  unsigned short* wqb   = ctxbf + NQKV;
  unsigned short* wkb   = wqb + NW;
  unsigned short* wvb   = wkb + NW;
  unsigned short* wob   = wvb + NW;
  int* mtop = (int*)(wob + NW);                        // 32*40

  const int CVT_BIG = (int)(NQKV / 4 / 256);           // 4096 blocks
  const int CVT_W   = (int)(NW   / 4 / 256);           // 256 blocks
  const int GEMM_BLOCKS = (M8 / 64) * (EE / 32) / 8;   // 256

  // f32 -> bf16 pre-conversion (once per buffer)
  cvt_bf16<<<CVT_BIG, 256, 0, stream>>>(query, xq);
  cvt_bf16<<<CVT_BIG, 256, 0, stream>>>(key,   xk);
  cvt_bf16<<<CVT_BIG, 256, 0, stream>>>(value, xv);
  cvt_bf16<<<CVT_W,   256, 0, stream>>>(Wq, wqb);
  cvt_bf16<<<CVT_W,   256, 0, stream>>>(Wk, wkb);
  cvt_bf16<<<CVT_W,   256, 0, stream>>>(Wv, wvb);
  cvt_bf16<<<CVT_W,   256, 0, stream>>>(Wo, wob);

  gemm_bf16<<<GEMM_BLOCKS, 256, 0, stream>>>(xq, wqb, bq, q,  nullptr);
  gemm_bf16<<<GEMM_BLOCKS, 256, 0, stream>>>(xk, wkb, bk, kf, kbf);
  gemm_bf16<<<GEMM_BLOCKS, 256, 0, stream>>>(xv, wvb, bv, vf, vbf);

  probe_kernel<<<BB * HH * LL / 8, 256, 0, stream>>>(q, kf, idxs, Mbuf);
  topk_kernel<<<BB * HH, 256, 0, stream>>>(Mbuf, mtop);
  vmean_kernel<<<BB * HH, 256, 0, stream>>>(vf, vmean);
  fill_ctx<<<(int)(NQKV / 256), 256, 0, stream>>>(ctx, vmean);
  gather_qtop<<<BB * HH * UP * DH / 256, 256, 0, stream>>>(q, mtop, qtop);

  scores_kernel<<<BB * HH * 384 / 8, 256, 0, stream>>>(qtop, kbf, S);
  softmax_kernel<<<BB * HH * UU, 256, 0, stream>>>(S, pbf);
  pv_kernel<<<BB * HH * 12 / 8, 256, 0, stream>>>(pbf, vbf, mtop, ctx);

  cvt_bf16<<<CVT_BIG, 256, 0, stream>>>(ctx, ctxbf);
  gemm_bf16<<<GEMM_BLOCKS, 256, 0, stream>>>(ctxbf, wob, bo, out, nullptr);
}
